// LowRankAttention_27754078666961
// MI455X (gfx1250) — matrix operand order repaired and checked by reading, not compiled, not run on hardware
//
#include <hip/hip_runtime.h>

typedef __attribute__((ext_vector_type(16))) __bf16 v16bf;
typedef __attribute__((ext_vector_type(8)))  __bf16 v8bf;
typedef __attribute__((ext_vector_type(8)))  float  v8f;

#define BB 4
#define NN 4096
#define DD 1024
#define RR 32

static __device__ __forceinline__ v16bf cat16(v8bf a, v8bf b) {
  return __builtin_shufflevector(a, b, 0,1,2,3,4,5,6,7,8,9,10,11,12,13,14,15);
}

// ---------------- Kernel 1: u = (X @ W) * scale, stored bf16 [B*N, R] ----------------
__global__ __launch_bounds__(256) void lra_proj(const float* __restrict__ X,
                                                const float* __restrict__ W,
                                                __bf16* __restrict__ U,
                                                float scale) {
  const int r = threadIdx.x & 31;
  const size_t row = (size_t)blockIdx.x * 8 + (threadIdx.x >> 5);
  const float* xr = X + row * DD;
  float acc = 0.0f;
#pragma unroll 8
  for (int d = 0; d < DD; ++d)
    acc = fmaf(xr[d], W[d * RR + r], acc);
  U[row * RR + r] = (__bf16)(acc * scale);
}

// ---------------- Kernel 2: vT[b][d][n] = bf16(v[b][n][d]) ----------------
__global__ __launch_bounds__(256) void lra_transpose(const float* __restrict__ V,
                                                     __bf16* __restrict__ VT) {
  __shared__ float tile[32][33];
  const int b  = blockIdx.z;
  const int n0 = blockIdx.x * 32;
  const int d0 = blockIdx.y * 32;
  const int tx = threadIdx.x & 31;
  const int ty = threadIdx.x >> 5;
#pragma unroll
  for (int i = 0; i < 4; ++i) {
    int nrow = ty + i * 8;
    tile[nrow][tx] = V[((size_t)b * NN + n0 + nrow) * DD + d0 + tx];
  }
  __syncthreads();
#pragma unroll
  for (int i = 0; i < 4; ++i) {
    int drow = ty + i * 8;
    VT[((size_t)b * DD + d0 + drow) * NN + n0 + tx] = (__bf16)tile[tx][drow];
  }
}

// ---------------- Kernel 3a: row maxima of (scaled, log2-domain) scores ----------------
__global__ __launch_bounds__(256) void lra_rowmax(const __bf16* __restrict__ u_bf,
                                                  const __bf16* __restrict__ vp_bf,
                                                  float* __restrict__ m_out) {
  const int b     = blockIdx.y;
  const int nb    = blockIdx.x * 16;
  const int wave  = threadIdx.x >> 5;
  const int lane  = threadIdx.x & 31;
  const int laneN = lane & 15;
  const int hi    = lane >> 4;

  __shared__ float lm[8][16];

  const __bf16* urow = u_bf + ((size_t)(b * NN + nb + laneN)) * RR;
  v16bf uA = cat16(*(const v8bf*)(urow + hi * 16),
                   *(const v8bf*)(urow + hi * 16 + 8));

  const v8f zc = {0.f, 0.f, 0.f, 0.f, 0.f, 0.f, 0.f, 0.f};
  const __bf16* vpb = vp_bf + (size_t)b * NN * RR;

  float mx[8];
#pragma unroll
  for (int j = 0; j < 8; ++j) mx[j] = -INFINITY;

  for (int kc = wave * 32; kc < NN; kc += 256) {
    const __bf16* r0 = vpb + (size_t)(kc + 2 * laneN) * RR + hi * 16;
    v16bf vb0 = cat16(*(const v8bf*)r0, *(const v8bf*)(r0 + 8));
    v16bf vb1 = cat16(*(const v8bf*)(r0 + RR), *(const v8bf*)(r0 + RR + 8));
    v8f s0 = __builtin_amdgcn_wmma_f32_16x16x32_bf16(false, uA, false, vb0,
                                                     (short)0, zc, false, false);
    v8f s1 = __builtin_amdgcn_wmma_f32_16x16x32_bf16(false, uA, false, vb1,
                                                     (short)0, zc, false, false);
#pragma unroll
    for (int j = 0; j < 8; ++j)
      mx[j] = fmaxf(mx[j], fmaxf(s0[j], s1[j]));
  }
#pragma unroll
  for (int j = 0; j < 8; ++j) {
    float v = mx[j];
    v = fmaxf(v, __shfl_xor(v, 1, 32));
    v = fmaxf(v, __shfl_xor(v, 2, 32));
    v = fmaxf(v, __shfl_xor(v, 4, 32));
    v = fmaxf(v, __shfl_xor(v, 8, 32));
    mx[j] = v;
  }
  if (laneN == 0) {
#pragma unroll
    for (int j = 0; j < 8; ++j) lm[wave][j + 8 * hi] = mx[j];
  }
  __syncthreads();
  if (threadIdx.x < 16) {
    float v = lm[0][threadIdx.x];
#pragma unroll
    for (int w = 1; w < 8; ++w) v = fmaxf(v, lm[w][threadIdx.x]);
    m_out[b * NN + nb + threadIdx.x] = v;
  }
}

// ---------------- Kernel 3b: attention, 32 query rows/block, wave-parallel P ----------------
// Grid: (N/32, B). 256 threads = 8 waves; wave w owns D columns [w*128, w*128+128).
// Per 256-key round: wave w produces P (2 row tiles) for chunk kg + w*32 into LDS
// (double-buffered), one barrier, then all waves consume all 8 chunks; each loaded
// vB tile feeds TWO O-WMMAs (one per row tile) -> 1:1 wmma:vmem.
__global__ __launch_bounds__(256) void lra_attn(const __bf16* __restrict__ u_bf,
                                                const __bf16* __restrict__ vp_bf,
                                                const __bf16* __restrict__ vT,
                                                const float* __restrict__ m_g,
                                                float* __restrict__ out) {
  const int b     = blockIdx.y;
  const int nb    = blockIdx.x * 32;
  const int wave  = threadIdx.x >> 5;
  const int lane  = threadIdx.x & 31;
  const int laneN = lane & 15;
  const int hi    = lane >> 4;

  __shared__ __align__(64) __bf16 pS[2][8][2][16 * 32];  // double-buffered P: 32KB
  __shared__ float lm[8][2][16];

  const __bf16* urow0 = u_bf + ((size_t)(b * NN + nb + laneN)) * RR;
  const __bf16* urow1 = urow0 + 16 * RR;
  v16bf uA0 = cat16(*(const v8bf*)(urow0 + hi * 16),
                    *(const v8bf*)(urow0 + hi * 16 + 8));
  v16bf uA1 = cat16(*(const v8bf*)(urow1 + hi * 16),
                    *(const v8bf*)(urow1 + hi * 16 + 8));

  v8f acc0[8], acc1[8];
  const v8f zc = {0.f, 0.f, 0.f, 0.f, 0.f, 0.f, 0.f, 0.f};
#pragma unroll
  for (int t = 0; t < 8; ++t) { acc0[t] = zc; acc1[t] = zc; }

  float m_i[2][8], l_i[2][8];
#pragma unroll
  for (int rt = 0; rt < 2; ++rt)
#pragma unroll
    for (int j = 0; j < 8; ++j) {
      m_i[rt][j] = m_g[b * NN + nb + rt * 16 + j + 8 * hi];
      l_i[rt][j] = 0.0f;
    }

  const int dbase = wave * 128;
  const __bf16* vpw = vp_bf + (size_t)b * NN * RR
                    + (size_t)(wave * 32 + 2 * laneN) * RR + hi * 16;
  const __bf16* vbase = vT + (size_t)b * DD * NN
                      + (size_t)(dbase + laneN) * NN + hi * 16;

  for (int kg = 0; kg < NN; kg += 256) {
    const int buf = (kg >> 8) & 1;

    // ---- produce P for this wave's chunk, both row tiles ----
    const __bf16* r0 = vpw + (size_t)kg * RR;
    v16bf vb0 = cat16(*(const v8bf*)r0, *(const v8bf*)(r0 + 8));
    v16bf vb1 = cat16(*(const v8bf*)(r0 + RR), *(const v8bf*)(r0 + RR + 8));
    v8f s0a = __builtin_amdgcn_wmma_f32_16x16x32_bf16(false, uA0, false, vb0,
                                                      (short)0, zc, false, false);
    v8f s1a = __builtin_amdgcn_wmma_f32_16x16x32_bf16(false, uA0, false, vb1,
                                                      (short)0, zc, false, false);
    v8f s0b = __builtin_amdgcn_wmma_f32_16x16x32_bf16(false, uA1, false, vb0,
                                                      (short)0, zc, false, false);
    v8f s1b = __builtin_amdgcn_wmma_f32_16x16x32_bf16(false, uA1, false, vb1,
                                                      (short)0, zc, false, false);
    __bf16* pw0 = &pS[buf][wave][0][0];
    __bf16* pw1 = &pS[buf][wave][1][0];
#pragma unroll
    for (int j = 0; j < 8; ++j) {
      const int row = j + hi * 8;
      {
        float p0 = __builtin_amdgcn_exp2f(s0a[j] - m_i[0][j]);
        float p1 = __builtin_amdgcn_exp2f(s1a[j] - m_i[0][j]);
        l_i[0][j] += p0 + p1;
        union { __bf16 h[2]; unsigned int u32; } pk;
        pk.h[0] = (__bf16)p0;
        pk.h[1] = (__bf16)p1;
        *(unsigned int*)&pw0[row * 32 + 2 * laneN] = pk.u32;
      }
      {
        float p0 = __builtin_amdgcn_exp2f(s0b[j] - m_i[1][j]);
        float p1 = __builtin_amdgcn_exp2f(s1b[j] - m_i[1][j]);
        l_i[1][j] += p0 + p1;
        union { __bf16 h[2]; unsigned int u32; } pk;
        pk.h[0] = (__bf16)p0;
        pk.h[1] = (__bf16)p1;
        *(unsigned int*)&pw1[row * 32 + 2 * laneN] = pk.u32;
      }
    }

    __syncthreads();   // single barrier per round (double buffering covers WAR)

    // ---- consume all 8 chunks; each vB feeds both row tiles ----
    const __bf16* vround = vbase + kg;
#pragma unroll
    for (int c = 0; c < 8; ++c) {
      const __bf16* pr0 = &pS[buf][c][0][laneN * 32];
      const __bf16* pr1 = &pS[buf][c][1][laneN * 32];
      v16bf pA0 = cat16(*(const v8bf*)(pr0 + hi * 16),
                        *(const v8bf*)(pr0 + hi * 16 + 8));
      v16bf pA1 = cat16(*(const v8bf*)(pr1 + hi * 16),
                        *(const v8bf*)(pr1 + hi * 16 + 8));
#pragma unroll
      for (int t = 0; t < 8; ++t) {
        const __bf16* vr = vround + (size_t)t * 16 * NN + c * 32;
        v16bf vB = cat16(*(const v8bf*)vr, *(const v8bf*)(vr + 8));
        acc0[t] = __builtin_amdgcn_wmma_f32_16x16x32_bf16(false, pA0, false, vB,
                                                          (short)0, acc0[t], false, false);
        acc1[t] = __builtin_amdgcn_wmma_f32_16x16x32_bf16(false, pA1, false, vB,
                                                          (short)0, acc1[t], false, false);
      }
    }
  }

  // ---- epilogue: combine per-wave partial row sums, scale, store ----
#pragma unroll
  for (int rt = 0; rt < 2; ++rt)
#pragma unroll
    for (int j = 0; j < 8; ++j) {
      float l = l_i[rt][j];
      l += __shfl_xor(l, 1, 32);
      l += __shfl_xor(l, 2, 32);
      l += __shfl_xor(l, 4, 32);
      l += __shfl_xor(l, 8, 32);
      if (laneN == 0) lm[wave][rt][j + 8 * hi] = l;
    }
  __syncthreads();
#pragma unroll
  for (int j = 0; j < 8; ++j) {
    const int row = j + 8 * hi;
    float tot0 = 0.0f, tot1 = 0.0f;
#pragma unroll
    for (int w = 0; w < 8; ++w) { tot0 += lm[w][0][row]; tot1 += lm[w][1][row]; }
    const float inv0 = 1.0f / tot0;
    const float inv1 = 1.0f / tot1;
#pragma unroll
    for (int t = 0; t < 8; ++t) {
      out[((size_t)(b * NN + nb + row)) * DD + dbase + t * 16 + laneN]      = acc0[t][j] * inv0;
      out[((size_t)(b * NN + nb + 16 + row)) * DD + dbase + t * 16 + laneN] = acc1[t][j] * inv1;
    }
  }
}

extern "C" void kernel_launch(void* const* d_in, const int* in_sizes, int n_in,
                              void* d_out, int out_size, void* d_ws, size_t ws_size,
                              hipStream_t stream) {
  const float* q  = (const float*)d_in[0];
  const float* k  = (const float*)d_in[1];
  const float* v  = (const float*)d_in[2];
  const float* Wu = (const float*)d_in[3];
  const float* Wv = (const float*)d_in[4];
  float* out = (float*)d_out;

  // workspace layout: u_bf (1MB) | vp_bf (1MB) | vT (32MB) | m (64KB)
  __bf16* u_bf  = (__bf16*)d_ws;
  __bf16* vp_bf = u_bf  + (size_t)BB * NN * RR;
  __bf16* vT    = vp_bf + (size_t)BB * NN * RR;
  float*  m_g   = (float*)(vT + (size_t)BB * DD * NN);

  // scale = (1/sqrt(R)) * log2(e): scores in log2 domain -> exp2 is one trans op.
  const float scale = 0.17677669529663687f * 1.4426950408889634f;

  lra_proj<<<dim3(BB * NN / 8), 256, 0, stream>>>(q, Wu, u_bf, scale);
  lra_proj<<<dim3(BB * NN / 8), 256, 0, stream>>>(k, Wv, vp_bf, 1.0f);
  lra_transpose<<<dim3(NN / 32, DD / 32, BB), 256, 0, stream>>>(v, vT);
  lra_rowmax<<<dim3(NN / 16, BB), 256, 0, stream>>>(u_bf, vp_bf, m_g);
  lra_attn<<<dim3(NN / 32, BB), 256, 0, stream>>>(u_bf, vp_bf, vT, m_g, out);
}
